// HungarianMatcher_62508954026071
// MI455X (gfx1250) — matrix operand order repaired and checked by reading, not compiled, not run on hardware
//
#include <hip/hip_runtime.h>
#include <hip/hip_bf16.h>
#include <math.h>

typedef __attribute__((ext_vector_type(16))) _Float16 v16h;
typedef __attribute__((ext_vector_type(8)))  _Float16 v8h;
typedef __attribute__((ext_vector_type(8)))  float    v8f;

#define BS   32
#define NQ   900
#define NT   50
#define NK   80
#define QPAD 912      // 57 tiles of 16
#define KPAD 96       // 3 k-steps of 32
#define LDC  912      // padded leading dim of LSA cost rows

// ---------------------------------------------------------------------------
// Kernel 1: focal-loss class-cost table P[b][q][k] = pos - neg, stored as a
// two-term f16 split (Phi + Plo ~= f32 value) so the one-hot WMMA gather is
// f32-accurate. Padded regions (q>=900, k>=80) are zero-filled.
// ---------------------------------------------------------------------------
__global__ void compute_p_kernel(const float* __restrict__ logits,
                                 _Float16* __restrict__ Phi,
                                 _Float16* __restrict__ Plo,
                                 int total) {
  int idx = blockIdx.x * blockDim.x + threadIdx.x;
  if (idx >= total) return;
  int k = idx % KPAD;
  int q = (idx / KPAD) % QPAD;
  int b = idx / (KPAD * QPAD);
  float val = 0.0f;
  if (k < NK && q < NQ) {
    float x  = logits[((size_t)b * NQ + q) * NK + k];
    float p  = 1.0f / (1.0f + expf(-x));
    float om = 1.0f - p;
    float neg = 0.75f * p * p * (-logf(om + 1e-8f));
    float pos = 0.25f * om * om * (-logf(p + 1e-8f));
    val = pos - neg;
  }
  _Float16 hi = (_Float16)val;
  _Float16 lo = (_Float16)(val - (float)hi);
  Phi[idx] = hi;
  Plo[idx] = lo;
}

// ---------------------------------------------------------------------------
// Kernel 2: one wave per 16x16 tile. cost_class tile = P @ one-hot(tgt_ids)
// via chained v_wmma_f32_16x16x32_f16 (hi + lo split => exact selection).
// Epilogue fuses the L1 bbox cost and writes C[b][t][q] (t-major for LSA).
// ---------------------------------------------------------------------------
__global__ __launch_bounds__(32) void gemm_cost_kernel(
    const _Float16* __restrict__ Phi,
    const _Float16* __restrict__ Plo,
    const int*   __restrict__ labels,
    const float* __restrict__ pboxes,
    const float* __restrict__ tboxes,
    float* __restrict__ C) {
  const int b     = blockIdx.z;
  const int qbase = blockIdx.x * 16;
  const int tbase = blockIdx.y * 16;
  const int lane  = threadIdx.x;
  const int hf    = lane >> 4;   // lane half
  const int l16   = lane & 15;

  const int q_row = qbase + l16;                       // A-matrix row (per lane)
  const size_t rowoff = ((size_t)b * QPAD + q_row) * KPAD;
  const _Float16* arow_hi = Phi + rowoff;
  const _Float16* arow_lo = Plo + rowoff;

  const int t  = tbase + l16;                          // B-matrix column (per lane)
  const int id = (t < NT) ? labels[b * NT + t] : -1;

  v8f acc = {0.f, 0.f, 0.f, 0.f, 0.f, 0.f, 0.f, 0.f};
#pragma unroll
  for (int ks = 0; ks < 3; ++ks) {
    const int k0 = ks * 32;
    // A tile: lanes 0-15 hold K k0+0..7 / k0+16..23; lanes 16-31 K k0+8..15 / k0+24..31
    v8h hlo = *(const v8h*)(arow_hi + k0 + hf * 8);
    v8h hhi = *(const v8h*)(arow_hi + k0 + 16 + hf * 8);
    v8h llo = *(const v8h*)(arow_lo + k0 + hf * 8);
    v8h lhi = *(const v8h*)(arow_lo + k0 + 16 + hf * 8);
    v16h a_hi, a_lo;
#pragma unroll
    for (int i = 0; i < 8; ++i) {
      a_hi[i] = hlo[i]; a_hi[8 + i] = hhi[i];
      a_lo[i] = llo[i]; a_lo[8 + i] = lhi[i];
    }
    // B tile (one-hot of target ids): lane = N, lane half selects K 0-7,16-23 / 8-15,24-31
    const int kb = k0 + hf * 8;
    v16h bm;
#pragma unroll
    for (int i = 0; i < 16; ++i)
      bm[i] = (id == kb + i + (i >= 8 ? 8 : 0)) ? (_Float16)1.0f : (_Float16)0.0f;

    acc = __builtin_amdgcn_wmma_f32_16x16x32_f16(false, a_hi, false, bm,
                                                 (short)0, acc, false, false);
    acc = __builtin_amdgcn_wmma_f32_16x16x32_f16(false, a_lo, false, bm,
                                                 (short)0, acc, false, false);
  }

  // Epilogue: D layout -> lane holds 8 contiguous q at (qbase + hf*8), col t.
  if (t < NT) {
    const float* tb = tboxes + ((size_t)b * NT + t) * 4;
    const float tb0 = tb[0], tb1 = tb[1], tb2 = tb[2], tb3 = tb[3];
    float* crow = C + ((size_t)b * NT + t) * LDC;
#pragma unroll
    for (int j = 0; j < 8; ++j) {
      const int q = qbase + hf * 8 + j;
      float l1 = 0.0f;
      if (q < NQ) {
        const float* pb = pboxes + ((size_t)b * NQ + q) * 4;
        l1 = fabsf(pb[0] - tb0) + fabsf(pb[1] - tb1) +
             fabsf(pb[2] - tb2) + fabsf(pb[3] - tb3);
      }
      crow[q] = 2.0f * acc[j] + 0.25f * l1;
    }
  }
}

// ---------------------------------------------------------------------------
// Kernel 3: Jonker-Volgenant shortest-augmenting-path LSA, one workgroup per
// batch (8 wave32s). The 900-wide scans/argmin/dual-updates are parallel over
// 256 threads; path bookkeeping is done by thread 0 in LDS. Matches the
// reference's tie-break (lowest column index among equal minima).
// ---------------------------------------------------------------------------
__global__ __launch_bounds__(256) void lsa_kernel(
    const float* __restrict__ C,
    const float* __restrict__ logits,
    const float* __restrict__ pboxes,
    const float* __restrict__ tboxes,
    const int*   __restrict__ labels,
    float* __restrict__ out,         // d_out: 3200 index floats + 2 scalars
    float* __restrict__ partials) {  // [32][2] per-batch means
  __shared__ float v_[NQ + 1];
  __shared__ float minv[NQ + 1];
  __shared__ int   p_[NQ + 1];
  __shared__ int   way[NQ + 1];
  __shared__ unsigned char used[NQ + 1];
  __shared__ float u_[NT + 1];
  __shared__ float rval[256];
  __shared__ int   ridx[256];
  __shared__ int   j0sh, i0sh;
  __shared__ float dsh;
  __shared__ int   qi_s[NT], ti_s[NT];
  __shared__ float csum[256], bsum[256];

  const int b   = blockIdx.x;
  const int tid = threadIdx.x;
  const float* Cb = C + (size_t)b * NT * LDC;

  for (int j = tid; j <= NQ; j += 256) { v_[j] = 0.0f; p_[j] = 0; way[j] = 0; }
  if (tid <= NT) u_[tid] = 0.0f;
  __syncthreads();

  for (int i = 1; i <= NT; ++i) {
    if (tid == 0) { p_[0] = i; j0sh = 0; }
    for (int j = tid; j <= NQ; j += 256) { minv[j] = 3.0e38f; used[j] = 0; }
    __syncthreads();
    while (true) {
      if (tid == 0) { used[j0sh] = 1; i0sh = p_[j0sh]; }
      __syncthreads();
      const int   j0  = j0sh;
      const int   i0  = i0sh;
      const float ui0 = u_[i0];
      const float* crow = Cb + (size_t)(i0 - 1) * LDC;

      float best = 3.0e38f; int bestj = 0x7fffffff;
      for (int j = tid + 1; j <= NQ; j += 256) {
        if (!used[j]) {
          float cur = crow[j - 1] - ui0 - v_[j];
          float mv  = minv[j];
          if (cur < mv) { mv = cur; minv[j] = cur; way[j] = j0; }
          if (mv < best || (mv == best && j < bestj)) { best = mv; bestj = j; }
        }
      }
      rval[tid] = best; ridx[tid] = bestj;
      __syncthreads();
      for (int s = 128; s > 0; s >>= 1) {
        if (tid < s) {
          float ov = rval[tid + s]; int oj = ridx[tid + s];
          if (ov < rval[tid] || (ov == rval[tid] && oj < ridx[tid])) {
            rval[tid] = ov; ridx[tid] = oj;
          }
        }
        __syncthreads();
      }
      if (tid == 0) { dsh = rval[0]; j0sh = ridx[0]; }
      __syncthreads();
      const float delta = dsh;
      const int   j1    = j0sh;
      for (int j = tid; j <= NQ; j += 256) {
        if (used[j]) { u_[p_[j]] += delta; v_[j] -= delta; }  // distinct rows
        else         { minv[j] -= delta; }
      }
      __syncthreads();
      if (p_[j1] == 0) break;   // uniform: shared read after barrier
    }
    if (tid == 0) {             // augment along stored path
      int j0 = j0sh;
      while (j0 != 0) { int jn = way[j0]; p_[j0] = p_[jn]; j0 = jn; }
    }
    __syncthreads();
  }

  // Extract (ascending columns, matching scipy/reference ordering).
  if (tid == 0) {
    int s = 0;
    for (int j = 1; j <= NQ; ++j)
      if (p_[j] != 0) { qi_s[s] = j - 1; ti_s[s] = p_[j] - 1; ++s; }
  }
  __syncthreads();

  // Indices + matched-cost partial sums (recomputed in f32 from raw inputs).
  float cl = 0.0f, bx = 0.0f;
  if (tid < NT) {
    const int q = qi_s[tid], t = ti_s[tid];
    out[b * 100 + tid]      = (float)q;
    out[b * 100 + 50 + tid] = (float)t;
    const int id = labels[b * NT + t];
    float x  = logits[((size_t)b * NQ + q) * NK + id];
    float p  = 1.0f / (1.0f + expf(-x));
    float om = 1.0f - p;
    float neg = 0.75f * p * p * (-logf(om + 1e-8f));
    float pos = 0.25f * om * om * (-logf(p + 1e-8f));
    cl = pos - neg;
    const float* pb = pboxes + ((size_t)b * NQ + q) * 4;
    const float* tb = tboxes + ((size_t)b * NT + t) * 4;
    bx = fabsf(pb[0] - tb[0]) + fabsf(pb[1] - tb[1]) +
         fabsf(pb[2] - tb[2]) + fabsf(pb[3] - tb[3]);
  }
  csum[tid] = cl; bsum[tid] = bx;
  __syncthreads();
  for (int s = 128; s > 0; s >>= 1) {
    if (tid < s) { csum[tid] += csum[tid + s]; bsum[tid] += bsum[tid + s]; }
    __syncthreads();
  }
  if (tid == 0) {
    partials[b * 2 + 0] = csum[0] / (float)NT;
    partials[b * 2 + 1] = bsum[0] / (float)NT;
  }
}

// ---------------------------------------------------------------------------
// Kernel 4: average the 32 per-batch means into the two output scalars.
// ---------------------------------------------------------------------------
__global__ void finalize_kernel(const float* __restrict__ partials,
                                float* __restrict__ out) {
  if (threadIdx.x == 0 && blockIdx.x == 0) {
    float c = 0.0f, bx = 0.0f;
    for (int b = 0; b < BS; ++b) { c += partials[b * 2 + 0]; bx += partials[b * 2 + 1]; }
    out[BS * 100 + 0] = c  / (float)BS;
    out[BS * 100 + 1] = bx / (float)BS;
  }
}

extern "C" void kernel_launch(void* const* d_in, const int* in_sizes, int n_in,
                              void* d_out, int out_size, void* d_ws, size_t ws_size,
                              hipStream_t stream) {
  const float* logits = (const float*)d_in[0];  // [32,900,80]
  const float* pboxes = (const float*)d_in[1];  // [32,900,4]
  const int*   labels = (const int*)d_in[2];    // [32,50]
  const float* tboxes = (const float*)d_in[3];  // [32,50,4]
  float* out = (float*)d_out;                   // 3202 floats

  // Workspace layout (all offsets 256B-aligned):
  //   Phi f16 [32][912][96]  : 5,603,328 B
  //   Plo f16 [32][912][96]  : 5,603,328 B
  //   C   f32 [32][50][912]  : 5,836,800 B
  //   partials f32 [32][2]   : 256 B
  char* ws = (char*)d_ws;
  const size_t planeElems = (size_t)BS * QPAD * KPAD;     // 2,801,664
  _Float16* Phi = (_Float16*)ws;
  _Float16* Plo = (_Float16*)(ws + planeElems * 2);
  float*    Cm  = (float*)(ws + planeElems * 4);
  float*    prt = (float*)(ws + planeElems * 4 + (size_t)BS * NT * LDC * 4);

  const int totalP = (int)planeElems;
  compute_p_kernel<<<(totalP + 255) / 256, 256, 0, stream>>>(logits, Phi, Plo, totalP);
  gemm_cost_kernel<<<dim3(QPAD / 16, 4, BS), 32, 0, stream>>>(Phi, Plo, labels,
                                                              pboxes, tboxes, Cm);
  lsa_kernel<<<BS, 256, 0, stream>>>(Cm, logits, pboxes, tboxes, labels, out, prt);
  finalize_kernel<<<1, 32, 0, stream>>>(prt, out);
}
